// Transf_15290083574072
// MI455X (gfx1250) — compile-verified, matrix operand order checked by reading, not run on hardware
//
#include <hip/hip_runtime.h>
#include <hip/hip_bf16.h>

typedef __attribute__((ext_vector_type(16))) __bf16 v16bf;
typedef __attribute__((ext_vector_type(8)))  __bf16 v8bf;
typedef __attribute__((ext_vector_type(8)))  float  v8f;
typedef __attribute__((ext_vector_type(2)))  float  v2f;

#define HAVE_WMMA_F32X4 __has_builtin(__builtin_amdgcn_wmma_f32_16x16x4_f32)

#define BATCH 2
#define SEQ   2048
#define DMODEL 32
#define FF    128
#define VOCAB 32000
#define NROWS (BATCH * SEQ)
#define LOG2E 1.4426950408889634f
#define INV_SQRT_D 0.17677669529663687f   // 1/sqrt(32)

__device__ __forceinline__ float wave_sum32(float v) {
#pragma unroll
  for (int m = 16; m >= 1; m >>= 1) v += __shfl_xor(v, m, 32);
  return v;
}

#define WMMA_BF16(A, B, C) \
  __builtin_amdgcn_wmma_f32_16x16x32_bf16(false, (A), false, (B), (short)0, (C), false, false)

// ---------------------------------------------------------------------------
// Kernel 1: h = emb[x]+pos ; hn = LN1(h) ; Q/K/V = hn @ W^T.
// One wave per token row (lane == channel). Q/K/V are emitted as bf16
// PRE-SWIZZLED into the exact CDNA5 WMMA register layouts:
//   Qs : A-layout 16-row tiles   [tile][lane(sl)][e]   (scaled by 1/sqrt(D)*log2e)
//   Ks : B-layout 16-key tiles   [tile][sl][e]
//   Vs : B-layout 32-key chunks  [chunk][d-half][sl][e]
// so the attention kernel loads operands with contiguous 32B vector loads.
// ---------------------------------------------------------------------------
__global__ void k_embed_ln_qkv(const int* __restrict__ x,
                               const float* __restrict__ emb,
                               const float* __restrict__ pos,
                               const float* __restrict__ Wq,
                               const float* __restrict__ Wk,
                               const float* __restrict__ Wv,
                               const float* __restrict__ g,
                               const float* __restrict__ bt,
                               float* __restrict__ h,
                               __bf16* __restrict__ Qs,
                               __bf16* __restrict__ Ks,
                               __bf16* __restrict__ Vs) {
  int lane = threadIdx.x & 31;
  int wid  = threadIdx.x >> 5;
  int rid  = blockIdx.x * 8 + wid;          // 0..4095 (b*SEQ + s)
  int s    = rid & (SEQ - 1);
  int tok  = x[rid];

  float hv  = emb[tok * DMODEL + lane] + pos[s * DMODEL + lane];
  float mu  = wave_sum32(hv) * (1.0f / 32.0f);
  float dc  = hv - mu;
  float var = wave_sum32(dc * dc) * (1.0f / 32.0f);
  float hn  = dc * rsqrtf(var + 1e-5f) * g[lane] + bt[lane];

  float q = 0.f, k = 0.f, v = 0.f;
#pragma unroll
  for (int d = 0; d < 32; ++d) {
    float t = __shfl(hn, d, 32);
    q += t * Wq[lane * DMODEL + d];
    k += t * Wk[lane * DMODEL + d];
    v += t * Wv[lane * DMODEL + d];
  }
  h[rid * DMODEL + lane] = hv;

  // 16-bit A/B layout inverse map for this lane's channel d (K-slot math)
  int d  = lane;
  int gq = (d >> 3) & 1;                       // half-wave group
  int eq = (d & 7) + ((d & 16) ? 8 : 0);       // element within v16bf
  size_t tb = (size_t)(rid >> 4) * 512;
  int me = rid & 15;
  Qs[tb + (size_t)(me + 16 * gq) * 16 + eq] = (__bf16)(q * (INV_SQRT_D * LOG2E));
  Ks[tb + (size_t)(me + 16 * gq) * 16 + eq] = (__bf16)k;

  int kk = rid & 31;                           // key position within 32-key chunk
  int gv = (kk >> 3) & 1;
  int ev = (kk & 7) + ((kk & 16) ? 8 : 0);
  Vs[(size_t)(rid >> 5) * 1024 + (size_t)(d >> 4) * 512 +
     (size_t)((d & 15) + 16 * gv) * 16 + ev] = (__bf16)v;
}

// ---------------------------------------------------------------------------
// Kernel 2: causal attention, split-K flash without running max (scores are
// O(1) here, so exp() cannot overflow and partial sums are additive).
// 4 waves per 16-query tile; row-sum L via a ones-column WMMA; partials
// combined through LDS. 5-6 WMMAs per 32-key chunk, minimal VALU.
// ---------------------------------------------------------------------------
template <bool MASKED>
__device__ __forceinline__ void attn_chunk(int c, int q0, int lane,
                                           const __bf16* __restrict__ Kb,
                                           const __bf16* __restrict__ Vb,
                                           const v16bf& aQ, const v16bf& bones,
                                           v8f& Olo, v8f& Ohi, v8f& La,
                                           __bf16 (&Pst)[16][32]) {
  int m  = lane & 15;
  int n  = m;
  int kb = (lane & 16) ? 8 : 0;
  int mo = (lane & 16) ? 8 : 0;

  const __bf16* kt = Kb + (size_t)c * 1024;
  v16bf bk0 = *(const v16bf*)(kt + lane * 16);
  v16bf bk1 = *(const v16bf*)(kt + 512 + lane * 16);
  v8f s0 = {}; v8f s1 = {};
  s0 = WMMA_BF16(aQ, bk0, s0);               // scores pre-scaled: s = qk/sqrt(D)*log2e
  s1 = WMMA_BF16(aQ, bk1, s1);

  int k0 = c * 32;
#pragma unroll
  for (int r = 0; r < 8; ++r) {
    float p0 = exp2f(s0[r]);
    float p1 = exp2f(s1[r]);
    if (MASKED) {
      int qg = q0 + r + mo;
      p0 = ((k0 + n)      <= qg) ? p0 : 0.0f;
      p1 = ((k0 + 16 + n) <= qg) ? p1 : 0.0f;
    }
    Pst[r + mo][n]      = (__bf16)p0;
    Pst[r + mo][n + 16] = (__bf16)p1;
  }
  asm volatile("s_wait_dscnt 0" ::: "memory");   // wave-local C->A relayout RAW

  const __bf16* prow = &Pst[m][0];
  v8bf plo = *(const v8bf*)(prow + kb);          // cols kb..kb+7
  v8bf phi = *(const v8bf*)(prow + kb + 16);     // cols kb+16..kb+23
  v16bf pa = __builtin_shufflevector(plo, phi, 0, 1, 2, 3, 4, 5, 6, 7,
                                     8, 9, 10, 11, 12, 13, 14, 15);

  const __bf16* vt = Vb + (size_t)c * 1024;
  v16bf bvlo = *(const v16bf*)(vt + lane * 16);
  v16bf bvhi = *(const v16bf*)(vt + 512 + lane * 16);
  Olo = WMMA_BF16(pa, bvlo, Olo);
  Ohi = WMMA_BF16(pa, bvhi, Ohi);
  La  = WMMA_BF16(pa, bones, La);                // row sums accumulate in col 0
}

__global__ void k_attn(const __bf16* __restrict__ Qs,
                       const __bf16* __restrict__ Ks,
                       const __bf16* __restrict__ Vs,
                       const float* __restrict__ h,
                       float* __restrict__ h2) {
  __shared__ __align__(16) __bf16 Pst[8][16][32];   // per-wave staging
  __shared__ float comb[8][3][8][32];               // split-K partials

  int lane = threadIdx.x & 31;
  int wid  = threadIdx.x >> 5;
  int part = wid & 3;                          // split-K part 0..3
  int t    = blockIdx.x * 2 + (wid >> 2);      // global 16-query tile 0..255
  int b    = t >> 7;
  int q0   = (t & 127) << 4;

  const __bf16* Kb = Ks + (size_t)b * 128 * 512;
  const __bf16* Vb = Vs + (size_t)b * 64 * 1024;

  v16bf aQ = *(const v16bf*)(Qs + (size_t)t * 512 + lane * 16);

  int n  = lane & 15;
  int mo = (lane & 16) ? 8 : 0;
  __bf16 onev = (n == 0) ? (__bf16)1.0f : (__bf16)0.0f;
  v16bf bones;
#pragma unroll
  for (int e = 0; e < 16; ++e) bones[e] = onev;

  v8f Olo = {}; v8f Ohi = {}; v8f La = {};
  int nch   = (q0 + 47) >> 5;                  // 32-key chunks in causal span
  int nfull = nch - 1;                         // exactly one straddles diagonal
  for (int c = part; c < nfull; c += 4)
    attn_chunk<false>(c, q0, lane, Kb, Vb, aQ, bones, Olo, Ohi, La, Pst[wid]);
  if (part == (nfull & 3))
    attn_chunk<true>(nfull, q0, lane, Kb, Vb, aQ, bones, Olo, Ohi, La, Pst[wid]);

#pragma unroll
  for (int r = 0; r < 8; ++r) {
    comb[wid][0][r][lane] = Olo[r];
    comb[wid][1][r][lane] = Ohi[r];
    comb[wid][2][r][lane] = La[r];
  }
  __syncthreads();

  if (part == 0) {                             // combine 4 parts + residual
#pragma unroll
    for (int r = 0; r < 8; ++r) {
      float o0 = 0.f, o1 = 0.f, l = 0.f;
#pragma unroll
      for (int p = 0; p < 4; ++p) {
        o0 += comb[wid + p][0][r][lane];
        o1 += comb[wid + p][1][r][lane];
        l  += comb[wid + p][2][r][lane & 16];  // L lives at col 0 of each half
      }
      float invL = 1.0f / l;
      int row = b * SEQ + q0 + r + mo;
      h2[row * DMODEL + n]      = o0 * invL + h[row * DMODEL + n];
      h2[row * DMODEL + n + 16] = o1 * invL + h[row * DMODEL + n + 16];
    }
  }
}

// ---------------------------------------------------------------------------
// Kernel 3: LN2 + FFN (+residual). One wave per row; per-wave LDS bounce.
// ---------------------------------------------------------------------------
__global__ void k_ffn(const float* __restrict__ h2,
                      const float* __restrict__ g,
                      const float* __restrict__ bt,
                      const float* __restrict__ W1,
                      const float* __restrict__ b1,
                      const float* __restrict__ W2,
                      const float* __restrict__ b2,
                      float* __restrict__ h3) {
  __shared__ float hns[8][32];
  __shared__ float ffs[8][128];
  int lane = threadIdx.x & 31;
  int wid  = threadIdx.x >> 5;
  int rid  = blockIdx.x * 8 + wid;

  float hv  = h2[rid * DMODEL + lane];
  float mu  = wave_sum32(hv) * (1.0f / 32.0f);
  float dc  = hv - mu;
  float var = wave_sum32(dc * dc) * (1.0f / 32.0f);
  float hn  = dc * rsqrtf(var + 1e-5f) * g[lane] + bt[lane];
  hns[wid][lane] = hn;
  asm volatile("s_wait_dscnt 0" ::: "memory");

#pragma unroll
  for (int jj = 0; jj < 4; ++jj) {
    int j = lane * 4 + jj;
    float acc = b1[j];
#pragma unroll
    for (int d = 0; d < 32; ++d) acc += hns[wid][d] * W1[j * DMODEL + d];
    ffs[wid][j] = fmaxf(acc, 0.0f);
  }
  asm volatile("s_wait_dscnt 0" ::: "memory");

  float acc = b2[lane];
#pragma unroll 8
  for (int j = 0; j < 128; ++j) acc += ffs[wid][j] * W2[lane * FF + j];
  h3[rid * DMODEL + lane] = acc + hv;
}

// ---------------------------------------------------------------------------
// Kernel 4: logits = h3 @ Wo^T ([4096,32]x[32,32000]) in full-precision
// f32 WMMA; bandwidth-bound on the 524 MB store -> non-temporal stores.
// ---------------------------------------------------------------------------
__global__ void k_logits(const float* __restrict__ h3,
                         const float* __restrict__ Wo,
                         float* __restrict__ out) {
  int lane = threadIdx.x & 31;
  int wid  = threadIdx.x >> 5;
  int col0 = (blockIdx.x * 8 + wid) * 16;
  int row0 = blockIdx.y * 16;
  int n  = lane & 15;
  int mo = (lane & 16) ? 8 : 0;
  v8f c = {};
#if HAVE_WMMA_F32X4
  int m  = lane & 15;
  int ko = (lane & 16) ? 2 : 0;
#pragma unroll
  for (int ks = 0; ks < 8; ++ks) {
    int kk = ks * 4 + ko;
    v2f a;  a[0] = h3[(row0 + m) * DMODEL + kk];
            a[1] = h3[(row0 + m) * DMODEL + kk + 1];
    v2f bb; bb[0] = Wo[(col0 + n) * DMODEL + kk];
            bb[1] = Wo[(col0 + n) * DMODEL + kk + 1];
    c = __builtin_amdgcn_wmma_f32_16x16x4_f32(false, a, false, bb, (short)0, c, false, false);
  }
#else
  int m  = lane & 15;
  int kb = (lane & 16) ? 8 : 0;
  v16bf a, bb;
#pragma unroll
  for (int e = 0; e < 16; ++e) {
    int kk = kb + e + (e >= 8 ? 8 : 0);
    a[e]  = (__bf16)h3[(row0 + m) * DMODEL + kk];
    bb[e] = (__bf16)Wo[(col0 + n) * DMODEL + kk];
  }
  c = WMMA_BF16(a, bb, c);
#endif
#pragma unroll
  for (int r = 0; r < 8; ++r) {
    __builtin_nontemporal_store(
        c[r], &out[(size_t)(row0 + r + mo) * VOCAB + col0 + n]);
  }
}

// ---------------------------------------------------------------------------
extern "C" void kernel_launch(void* const* d_in, const int* in_sizes, int n_in,
                              void* d_out, int out_size, void* d_ws, size_t ws_size,
                              hipStream_t stream) {
  (void)in_sizes; (void)n_in; (void)out_size; (void)ws_size;
  const int*   x    = (const int*)d_in[0];
  const float* emb  = (const float*)d_in[1];
  const float* pos  = (const float*)d_in[2];
  const float* Wq   = (const float*)d_in[3];
  const float* Wk   = (const float*)d_in[4];
  const float* Wv   = (const float*)d_in[5];
  const float* ln1g = (const float*)d_in[6];
  const float* ln1b = (const float*)d_in[7];
  const float* ln2g = (const float*)d_in[8];
  const float* ln2b = (const float*)d_in[9];
  const float* W1   = (const float*)d_in[10];
  const float* b1   = (const float*)d_in[11];
  const float* W2   = (const float*)d_in[12];
  const float* b2   = (const float*)d_in[13];
  const float* Wo   = (const float*)d_in[14];

  char* ws = (char*)d_ws;
  const size_t SZF = (size_t)NROWS * DMODEL * sizeof(float);   // 512 KB
  const size_t SZB = (size_t)NROWS * DMODEL * sizeof(__bf16);  // 256 KB
  float*  h  = (float*)(ws);
  float*  h2 = (float*)(ws + SZF);
  float*  h3 = (float*)(ws + 2 * SZF);
  __bf16* Qs = (__bf16*)(ws + 3 * SZF);
  __bf16* Ks = (__bf16*)(ws + 3 * SZF + SZB);
  __bf16* Vs = (__bf16*)(ws + 3 * SZF + 2 * SZB);

  k_embed_ln_qkv<<<NROWS / 8, 256, 0, stream>>>(x, emb, pos, Wq, Wk, Wv,
                                                ln1g, ln1b, h, Qs, Ks, Vs);
  k_attn<<<(BATCH * (SEQ / 16)) / 2, 256, 0, stream>>>(Qs, Ks, Vs, h, h2);
  k_ffn<<<NROWS / 8, 256, 0, stream>>>(h2, ln2g, ln2b, W1, b1, W2, b2, h3);
  dim3 g4(VOCAB / (16 * 8), NROWS / 16);
  k_logits<<<g4, 256, 0, stream>>>(h3, Wo, (float*)d_out);
}
